// GraphVAEncoder_Link_67362267070872
// MI455X (gfx1250) — compile-verified, matrix operand order checked by reading, not run on hardware
//
#include <hip/hip_runtime.h>

typedef __attribute__((ext_vector_type(16))) _Float16 v16h;
typedef __attribute__((ext_vector_type(8)))  _Float16 v8h;
typedef __attribute__((ext_vector_type(8)))  float    v8f;

#define HIDD 128

// ---------------- degree / normalization ----------------

__global__ void k_deg_init(float* __restrict__ deg, int n) {
  int i = blockIdx.x * blockDim.x + threadIdx.x;
  if (i < n) deg[i] = 1.0f;  // self-loop weight 1
}

__global__ void k_deg_edges(const int* __restrict__ dst, const float* __restrict__ ew,
                            float* __restrict__ deg, int E) {
  int e = blockIdx.x * blockDim.x + threadIdx.x;
  if (e < E) atomicAdd(&deg[dst[e]], ew[e]);
}

__global__ void k_dinv(const float* __restrict__ deg, float* __restrict__ dinv, int n) {
  int i = blockIdx.x * blockDim.x + threadIdx.x;
  if (i < n) {
    float d = deg[i];
    dinv[i] = (d > 0.0f) ? rsqrtf(d) : 0.0f;
  }
}

// ---------------- f16 conversion / weight transpose ----------------

__global__ void k_cvt_f16(const float* __restrict__ src, _Float16* __restrict__ dst, int n) {
  int i = blockIdx.x * blockDim.x + threadIdx.x;
  if (i < n) dst[i] = (_Float16)src[i];
}

// W: [K][Nw] row-major -> Wt: [Nw][K] (column-major W), f16
__global__ void k_transpose_cvt(const float* __restrict__ W, _Float16* __restrict__ Wt,
                                int K, int Nw) {
  int i = blockIdx.x * blockDim.x + threadIdx.x;
  if (i < K * Nw) {
    int k = i / Nw, n = i % Nw;
    Wt[(size_t)n * K + k] = (_Float16)W[i];
  }
}

// ---------------- WMMA GEMM: C[M][128] = A[M][K] @ Bt[128][K]^T ----------------
// One block = 8 waves = one 16-row M tile; wave w computes N tile w (N=128).
// M % 16 == 0, K % 32 == 0 guaranteed by problem sizes.

__global__ void __launch_bounds__(256)
k_gemm_wmma(const _Float16* __restrict__ A, const _Float16* __restrict__ Bt,
            float* __restrict__ C, int K) {
  const int lane  = threadIdx.x & 31;
  const int wave  = threadIdx.x >> 5;      // 0..7 -> N tile
  const int mtile = blockIdx.x;

  const int ksel = (lane >> 4) << 3;       // lanes 0-15: K {0..7,16..23}; 16-31: {8..15,24..31}
  const int row  = mtile * 16 + (lane & 15);
  const int col  = wave  * 16 + (lane & 15);

  const _Float16* ap = A  + (size_t)row * K + ksel;
  const _Float16* bp = Bt + (size_t)col * K + ksel;

  v8f c = {};
  for (int k0 = 0; k0 < K; k0 += 32) {
    v8h alo = *(const v8h*)(ap + k0);
    v8h ahi = *(const v8h*)(ap + k0 + 16);
    v8h blo = *(const v8h*)(bp + k0);
    v8h bhi = *(const v8h*)(bp + k0 + 16);
    v16h a, b;
#pragma unroll
    for (int i = 0; i < 8; ++i) {
      a[i] = alo[i]; a[i + 8] = ahi[i];
      b[i] = blo[i]; b[i + 8] = bhi[i];
    }
    c = __builtin_amdgcn_wmma_f32_16x16x32_f16(
        /*neg_a=*/false, a, /*neg_b=*/false, b,
        /*c_mod=*/(short)0, c, /*reuse_a=*/false, /*reuse_b=*/false);
  }

  // D layout: lanes 0-15 -> (M=v, N=lane); lanes 16-31 -> (M=v+8, N=lane-16)
  const int m0 = mtile * 16 + ((lane >> 4) << 3);
  float* out = C + (size_t)m0 * HIDD + wave * 16 + (lane & 15);
#pragma unroll
  for (int v = 0; v < 8; ++v) out[(size_t)v * HIDD] = c[v];
}

// ---------------- aggregation ----------------

// agg[i][d] = dinv[i]^2 * h[i][d] + bias[d]   (self-loop term + bias)
__global__ void k_selfloop_init(const float* __restrict__ h, const float* __restrict__ dinv,
                                const float* __restrict__ bias, float* __restrict__ agg,
                                int total) {
  int i = blockIdx.x * blockDim.x + threadIdx.x;
  if (i < total) {
    int node = i >> 7;            // /128
    int d    = i & (HIDD - 1);
    float s = dinv[node];
    agg[i] = s * s * h[i] + bias[d];
  }
}

// edge-parallel scatter: one thread per (edge, feature)
__global__ void k_agg_edges(const int* __restrict__ src, const int* __restrict__ dst,
                            const float* __restrict__ ew, const float* __restrict__ dinv,
                            const float* __restrict__ h, float* __restrict__ agg, int E) {
  int t = blockIdx.x * blockDim.x + threadIdx.x;
  int e = t >> 7;
  if (e >= E) return;
  int d  = t & (HIDD - 1);
  int s  = src[e];
  int dn = dst[e];
  float norm = dinv[s] * ew[e] * dinv[dn];
  atomicAdd(&agg[(size_t)dn * HIDD + d], norm * h[(size_t)s * HIDD + d]);
}

__global__ void k_relu_cvt(const float* __restrict__ agg, _Float16* __restrict__ hf, int total) {
  int i = blockIdx.x * blockDim.x + threadIdx.x;
  if (i < total) {
    float v = agg[i];
    hf[i] = (_Float16)(v > 0.0f ? v : 0.0f);
  }
}

// ---------------- launcher ----------------

extern "C" void kernel_launch(void* const* d_in, const int* in_sizes, int n_in,
                              void* d_out, int out_size, void* d_ws, size_t ws_size,
                              hipStream_t stream) {
  const float* x  = (const float*)d_in[0];
  const int*   ei = (const int*)  d_in[1];
  const float* ew = (const float*)d_in[2];
  const float* W1 = (const float*)d_in[3];
  const float* b1 = (const float*)d_in[4];
  const float* W2 = (const float*)d_in[5];
  const float* b2 = (const float*)d_in[6];
  float* out = (float*)d_out;

  const int HID = in_sizes[4];              // 128
  const int IN  = in_sizes[3] / HID;        // 512
  const int N   = in_sizes[0] / IN;         // 65536
  const int E   = in_sizes[2];              // 1048576
  const int* srcp = ei;
  const int* dstp = ei + E;

  // workspace carve-up
  char* p = (char*)d_ws;
  float*    deg  = (float*)p;    p += (size_t)N * 4;
  float*    dinv = (float*)p;    p += (size_t)N * 4;
  _Float16* w1t  = (_Float16*)p; p += (size_t)IN * HID * 2;
  _Float16* w2t  = (_Float16*)p; p += (size_t)HID * HID * 2;
  _Float16* xh   = (_Float16*)p; p += (size_t)N * IN * 2;   // reused as h_f16 for layer 2
  float*    gout = (float*)p;    p += (size_t)N * HID * 4;  // GEMM output (both layers)
  float*    agg  = (float*)p;                               // layer-1 aggregation

  const int NT = 256;
  const int totNH = N * HID;

  // normalization
  k_deg_init <<<(N + NT - 1) / NT, NT, 0, stream>>>(deg, N);
  k_deg_edges<<<(E + NT - 1) / NT, NT, 0, stream>>>(dstp, ew, deg, E);
  k_dinv     <<<(N + NT - 1) / NT, NT, 0, stream>>>(deg, dinv, N);

  // precision conversion + weight transposes
  k_cvt_f16      <<<((size_t)N * IN + NT - 1) / NT, NT, 0, stream>>>(x, xh, N * IN);
  k_transpose_cvt<<<(IN * HID + NT - 1) / NT, NT, 0, stream>>>(W1, w1t, IN, HID);
  k_transpose_cvt<<<(HID * HID + NT - 1) / NT, NT, 0, stream>>>(W2, w2t, HID, HID);

  // layer 1: GEMM + aggregate + ReLU
  k_gemm_wmma<<<N / 16, 256, 0, stream>>>(xh, w1t, gout, IN);
  k_selfloop_init<<<(totNH + NT - 1) / NT, NT, 0, stream>>>(gout, dinv, b1, agg, totNH);
  {
    long long tot = (long long)E * HIDD;
    k_agg_edges<<<(unsigned)((tot + NT - 1) / NT), NT, 0, stream>>>(srcp, dstp, ew, dinv,
                                                                    gout, agg, E);
  }
  k_relu_cvt<<<(totNH + NT - 1) / NT, NT, 0, stream>>>(agg, xh, totNH);  // xh now holds h_f16

  // layer 2: GEMM + aggregate (directly into d_out)
  k_gemm_wmma<<<N / 16, 256, 0, stream>>>(xh, w2t, gout, HID);
  k_selfloop_init<<<(totNH + NT - 1) / NT, NT, 0, stream>>>(gout, dinv, b2, out, totNH);
  {
    long long tot = (long long)E * HIDD;
    k_agg_edges<<<(unsigned)((tot + NT - 1) / NT), NT, 0, stream>>>(srcp, dstp, ew, dinv,
                                                                    gout, out, E);
  }
}
